// SDT_GRU_Classifier_4741643895741
// MI455X (gfx1250) — matrix operand order repaired and checked by reading, not compiled, not run on hardware
//
#include <hip/hip_runtime.h>
#include <hip/hip_bf16.h>

// Problem constants (match reference)
#define BATCH 8192
#define NNODE 20
#define DMODEL 128
#define NHEAD 8
#define FFDIM 512
#define KCONV 3
#define LNEPS 1e-5f

typedef _Float16 half_t;
typedef half_t v8h  __attribute__((ext_vector_type(8)));
typedef half_t v16h __attribute__((ext_vector_type(16)));
typedef float  v8f  __attribute__((ext_vector_type(8)));

// ---------------------------------------------------------------------------
// WMMA helpers (CDNA5 wave32, V_WMMA_F32_16X16X32_F16)
// ---------------------------------------------------------------------------
__device__ __forceinline__ v16h cat8(v8h a, v8h b) {
  return __builtin_shufflevector(a, b, 0,1,2,3,4,5,6,7,8,9,10,11,12,13,14,15);
}

__device__ __forceinline__ v8f wmma16(v16h a, v16h b, v8f c) {
  // (neg_a, A, neg_b, B, c_mod, C, reuse_a, reuse_b)
  return __builtin_amdgcn_wmma_f32_16x16x32_f16(false, a, false, b, (short)0, c,
                                                false, false);
}

// A-matrix 16x32 f16 layout (ISA 7.12.2): lanes 0-15 row M=lane, K = {0..7,16..23};
// lanes 16-31 row M=lane-16, K = {8..15,24..31}. Two 16B LDS loads per fragment.
__device__ __forceinline__ v16h load_a_frag(const half_t* A, int lda, int mbase,
                                            int kc, int lane) {
  int row = mbase + (lane & 15);
  int k0  = kc * 32 + ((lane >> 4) << 3);
  const half_t* p = A + row * lda + k0;
  v8h lo = *(const v8h*)p;
  v8h hi = *(const v8h*)(p + 16);
  return cat8(lo, hi);
}

// B-matrix 32x16 f16 layout: lanes 0-15 column N=lane, K = {0..7,16..23};
// lanes 16-31 column N=lane-16, K = {8..15,24..31}. Weights packed [n][k] -> two 16B reads.
__device__ __forceinline__ v16h load_b_frag(const half_t* Bp, int ldb, int nt,
                                            int kc, int lane) {
  int col = nt * 16 + (lane & 15);
  int k0  = kc * 32 + ((lane >> 4) << 3);
  const half_t* p = Bp + col * ldb + k0;
  v8h lo = *(const v8h*)p;
  v8h hi = *(const v8h*)(p + 16);
  return cat8(lo, hi);
}

// C/D 16x16 f32 layout: VGPR e, lanes 0-15 -> M=e, N=lane; lanes 16-31 -> M=e+8.
__device__ __forceinline__ void store_acc_f32(v8f acc, float* dst, int ldd,
                                              int mbase, int nbase,
                                              const float* bias, int lane) {
  int col = nbase + (lane & 15);
  int r0  = (lane >> 4) << 3;
  float bv = bias ? bias[col] : 0.0f;
#pragma unroll
  for (int e = 0; e < 8; ++e) {
    int m = mbase + r0 + e;
    if (m < NNODE) dst[m * ldd + col] = acc[e] + bv;
  }
}

// Generic C(20xN) += A(20x32*NKC) @ B: wave = (mt = wave&1, column = wave>>1);
// A-fragments loaded once into registers, reused across NT/4 n-tiles.
template <int NKC, int NT>
__device__ __forceinline__ void gemm_store_f32(const half_t* A, int lda,
                                               const half_t* Bp, float* dst,
                                               int ldd, const float* bias,
                                               int wave, int lane) {
  const int mt = wave & 1;
  const int c0 = wave >> 1;
  v16h a[NKC];
#pragma unroll
  for (int kc = 0; kc < NKC; ++kc) a[kc] = load_a_frag(A, lda, mt * 16, kc, lane);
#pragma unroll
  for (int j = 0; j < NT / 4; ++j) {
    int nt = c0 + j * 4;
    v8f acc = {};
#pragma unroll
    for (int kc = 0; kc < NKC; ++kc) {
      v16h bf = load_b_frag(Bp, 32 * NKC, nt, kc, lane);
      acc = wmma16(a[kc], bf, acc);
    }
    store_acc_f32(acc, dst, ldd, mt * 16, nt * 16, bias, lane);
  }
}

// ---------------------------------------------------------------------------
// wave32 reduction + LayerNorm over one 128-wide row (one wave per row)
// ---------------------------------------------------------------------------
__device__ __forceinline__ float wred(float v) {
#pragma unroll
  for (int m = 16; m >= 1; m >>= 1) v += __shfl_xor(v, m, 32);
  return v;
}

// res_inside: y = LN(src+res); else y = LN(src)+res  (conv projections)
__device__ __forceinline__ void layernorm_rows(const float* src, int src_ld,
                                               const float* res, int res_ld,
                                               bool res_inside,
                                               const float* w, const float* bb,
                                               half_t* d16, float* d32, float* dG,
                                               int wave, int lane) {
  for (int r = wave; r < NNODE; r += 8) {
    const float* sr = src + r * src_ld;
    const float* rr = res + r * res_ld;
    int c = lane * 4;
    float v[4], rv[4];
#pragma unroll
    for (int j = 0; j < 4; ++j) {
      rv[j] = rr[c + j];
      v[j]  = sr[c + j] + (res_inside ? rv[j] : 0.0f);
    }
    float mean = wred(v[0] + v[1] + v[2] + v[3]) * 0.0078125f;
    float q2 = 0.0f;
#pragma unroll
    for (int j = 0; j < 4; ++j) { float d = v[j] - mean; q2 += d * d; }
    float rs = rsqrtf(wred(q2) * 0.0078125f + LNEPS);
#pragma unroll
    for (int j = 0; j < 4; ++j) {
      float y = (v[j] - mean) * rs * w[c + j] + bb[c + j];
      if (!res_inside) y += rv[j];
      if (d16) d16[r * DMODEL + c + j] = (half_t)y;
      if (d32) d32[r * DMODEL + c + j] = y;
      if (dG)  dG[r * DMODEL + c + j]  = y;
    }
  }
}

// ---------------------------------------------------------------------------
// Weight packing kernels: fp32 -> f16, B laid out [n][k] (column-major for GEMM)
// ---------------------------------------------------------------------------
__global__ void pack_conv_k(const float* __restrict__ wq, const float* __restrict__ wk,
                            const float* __restrict__ wv, half_t* __restrict__ dst) {
  int idx = blockIdx.x * 256 + threadIdx.x;
  if (idx >= 3 * KCONV * DMODEL * DMODEL) return;
  int p  = idx / (KCONV * DMODEL * DMODEL);
  int r  = idx - p * (KCONV * DMODEL * DMODEL);
  int t  = r / (DMODEL * DMODEL);
  int r2 = r - t * (DMODEL * DMODEL);
  int o  = r2 >> 7;
  int i  = r2 & 127;
  const float* w = (p == 0) ? wq : ((p == 1) ? wk : wv);
  dst[idx] = (half_t)w[(o * DMODEL + i) * KCONV + t];   // (out,in,K) -> [p][t][o][i]
}

__global__ void pack_T_k(const float* __restrict__ in, half_t* __restrict__ out,
                         int kin, int nout) {
  int idx = blockIdx.x * 256 + threadIdx.x;
  if (idx >= kin * nout) return;
  int o = idx / kin;
  int i = idx - o * kin;
  out[idx] = (half_t)in[i * nout + o];   // (k,n) row-major -> [n][k]
}

// ---------------------------------------------------------------------------
// Fused encoder block: one workgroup (8 wave32) per batch element
// ---------------------------------------------------------------------------
// LDS arena (61952 B), single static array => arena base is LDS offset 0:
//   [0,10240)      tmp32: 20x128 f32 GEMM staging (conv / attn_out / ffn2 out)
//   [10240,18944)  pad16: 34x128 f16 conv halo -> later ctx16 / xr16 (32x128 f16)
//   [18944,29184)  x32  : 20x128 f32 (async-staged x; later overwritten by xr)
//   [29184,61952)  phase1: scores 8x20x20 f32 @+0, qkv16 3x20x128 f16 @+12800
//                  phase2: h16 32x512 f16 (FFN hidden)
#define SM_TMP   0
#define SM_PAD   10240
#define SM_X32   18944
#define SM_F     29184
#define SM_BYTES 61952

__global__ __launch_bounds__(256) void fused_encoder_k(
    const float* __restrict__ x,
    const half_t* __restrict__ wpack,   // [3][3][128][128]
    const half_t* __restrict__ wop,     // [128][128]
    const half_t* __restrict__ w1p,     // [512][128]
    const half_t* __restrict__ w2p,     // [128][512]
    const float* __restrict__ lnq_w, const float* __restrict__ lnq_b,
    const float* __restrict__ lnk_w, const float* __restrict__ lnk_b,
    const float* __restrict__ lnv_w, const float* __restrict__ lnv_b,
    const float* __restrict__ rel_table, const float* __restrict__ gbias,
    const float* __restrict__ alphap,
    const float* __restrict__ bo, const float* __restrict__ b1,
    const float* __restrict__ b2,
    const float* __restrict__ n1_w, const float* __restrict__ n1_b,
    const float* __restrict__ n2_w, const float* __restrict__ n2_b,
    float* __restrict__ out) {
  __shared__ __align__(16) unsigned char smem[SM_BYTES];
  float*  s_tmp = (float*)(smem + SM_TMP);
  half_t* s_pad = (half_t*)(smem + SM_PAD);  // conv halo, then ctx16, then xr16
  float*  s_x32 = (float*)(smem + SM_X32);   // x, then xr
  float*  s_sc  = (float*)(smem + SM_F);               // 8*20*20
  half_t* s_qkv = (half_t*)(smem + SM_F + 12800);      // 3*20*128
  half_t* s_h   = (half_t*)(smem + SM_F);              // 32*512 (phase 2)

  const int tid  = threadIdx.x;
  const int lane = tid & 31;
  const int wave = tid >> 5;
  const int b    = blockIdx.x;

  const float* xg   = x + (size_t)b * NNODE * DMODEL;
  float*       outg = out + (size_t)b * NNODE * DMODEL;

  // ---- async-stage x (20x128 f32 = 640 x 16B) into LDS @ SM_X32 ----
  // ASYNCcnt-tracked copy, no VGPR round trip (CDNA5 ISA).
  for (int idx = tid; idx < 640; idx += 256) {
    unsigned lds_addr = (unsigned)(SM_X32 + idx * 16);
    unsigned long long gaddr = (unsigned long long)(const void*)(xg + idx * 4);
    asm volatile("global_load_async_to_lds_b128 %0, %1, off"
                 :: "v"(lds_addr), "v"(gaddr) : "memory");
  }
  asm volatile("s_wait_asynccnt 0x0" ::: "memory");
  __syncthreads();

  // ---- build f16 conv-halo buffer from LDS x: pad[j] = x[j-1], zeros outside ----
  for (int idx = tid; idx < 34 * DMODEL; idx += 256) {
    int j = idx >> 7, i = idx & 127;
    float v = (j >= 1 && j <= NNODE) ? s_x32[(j - 1) * DMODEL + i] : 0.0f;
    s_pad[idx] = (half_t)v;
  }
  __syncthreads();

  // ---- conv QKV projections (WMMA, A-frags register-resident) + LN + residual ----
  for (int p = 0; p < 3; ++p) {
    {
      const int mt = wave & 1;
      const int c0 = wave >> 1;
      v16h a[KCONV * 4];
#pragma unroll
      for (int t = 0; t < KCONV; ++t)
#pragma unroll
        for (int kc = 0; kc < 4; ++kc)
          a[t * 4 + kc] = load_a_frag(s_pad, DMODEL, mt * 16 + t, kc, lane);
#pragma unroll
      for (int j = 0; j < 2; ++j) {
        int nt = c0 + j * 4;
        v8f acc = {};
#pragma unroll
        for (int t = 0; t < KCONV; ++t) {
          const half_t* wB = wpack + (size_t)(p * KCONV + t) * DMODEL * DMODEL;
#pragma unroll
          for (int kc = 0; kc < 4; ++kc) {
            v16h bf = load_b_frag(wB, DMODEL, nt, kc, lane);
            acc = wmma16(a[t * 4 + kc], bf, acc);
          }
        }
        store_acc_f32(acc, s_tmp, DMODEL, mt * 16, nt * 16, nullptr, lane);
      }
    }
    __syncthreads();
    const float* lw = (p == 0) ? lnq_w : ((p == 1) ? lnk_w : lnv_w);
    const float* lb = (p == 0) ? lnq_b : ((p == 1) ? lnk_b : lnv_b);
    layernorm_rows(s_tmp, DMODEL, s_x32, DMODEL, /*inside=*/false, lw, lb,
                   s_qkv + p * NNODE * DMODEL, nullptr, nullptr, wave, lane);
    __syncthreads();
  }

  // ---- attention scores + biases (VALU; tiny 20x20 per head) ----
  const half_t* s_q = s_qkv;
  const half_t* s_k = s_qkv + NNODE * DMODEL;
  const half_t* s_v = s_qkv + 2 * NNODE * DMODEL;
  float alpha_v = alphap[0];
  for (int idx = tid; idx < NHEAD * NNODE * NNODE; idx += 256) {
    int h = idx / (NNODE * NNODE);
    int r = idx - h * NNODE * NNODE;
    int n = r / NNODE, m = r - (r / NNODE) * NNODE;
    const half_t* qr = s_q + n * DMODEL + h * 16;
    const half_t* kr = s_k + m * DMODEL + h * 16;
    float s = 0.0f;
#pragma unroll
    for (int d = 0; d < 16; ++d) s += (float)qr[d] * (float)kr[d];
    s *= 0.25f;                                        // 1/sqrt(16)
    s += rel_table[(n - m + NNODE - 1) * NHEAD + h];
    s += gbias[(h * NNODE + n) * NNODE + m] * alpha_v;
    s_sc[(h * NNODE + n) * NNODE + m] = s;
  }
  __syncthreads();
  // softmax per (h,n)
  for (int r = tid; r < NHEAD * NNODE; r += 256) {
    float* row = s_sc + r * NNODE;
    float mx = row[0];
#pragma unroll
    for (int m = 1; m < NNODE; ++m) mx = fmaxf(mx, row[m]);
    float sum = 0.0f;
#pragma unroll
    for (int m = 0; m < NNODE; ++m) { float e = __expf(row[m] - mx); row[m] = e; sum += e; }
    float inv = 1.0f / sum;
#pragma unroll
    for (int m = 0; m < NNODE; ++m) row[m] *= inv;
  }
  __syncthreads();
  // ctx = attn @ V  -> ctx16 (reuses halo slot)
  half_t* s_ctx = s_pad;
  for (int idx = tid; idx < NNODE * DMODEL; idx += 256) {
    int n = idx >> 7, o = idx & 127, h = o >> 4;
    const float*  ar = s_sc + (h * NNODE + n) * NNODE;
    const half_t* vc = s_v + o;
    float s = 0.0f;
#pragma unroll
    for (int m = 0; m < NNODE; ++m) s += ar[m] * (float)vc[m * DMODEL];
    s_ctx[n * DMODEL + o] = (half_t)s;
  }
  __syncthreads();

  // ---- attn_out = ctx @ wo + bo (WMMA) -> tmp32 ----
  gemm_store_f32<4, 8>(s_ctx, DMODEL, wop, s_tmp, DMODEL, bo, wave, lane);
  __syncthreads();

  // ---- xr = LN(x + attn_out) -> x32 (overwrite) + xr16 (overlays ctx slot) ----
  half_t* s_xr16 = s_pad;
  layernorm_rows(s_tmp, DMODEL, s_x32, DMODEL, /*inside=*/true, n1_w, n1_b,
                 s_xr16, s_x32, nullptr, wave, lane);
  __syncthreads();

  // ---- FFN1: h = gelu(xr @ w1 + b1) (WMMA, N=512, A reused over 8 n-tiles) ----
  {
    const int mt = wave & 1;
    const int c0 = wave >> 1;
    v16h a[4];
#pragma unroll
    for (int kc = 0; kc < 4; ++kc) a[kc] = load_a_frag(s_xr16, DMODEL, mt * 16, kc, lane);
#pragma unroll
    for (int j = 0; j < 8; ++j) {
      int nt = c0 + j * 4;
      v8f acc = {};
#pragma unroll
      for (int kc = 0; kc < 4; ++kc) {
        v16h bf = load_b_frag(w1p, DMODEL, nt, kc, lane);
        acc = wmma16(a[kc], bf, acc);
      }
      int col = nt * 16 + (lane & 15);
      int r0  = (lane >> 4) << 3;
      float bv = b1[col];
#pragma unroll
      for (int e = 0; e < 8; ++e) {
        int m = mt * 16 + r0 + e;
        if (m < NNODE) {
          float t = acc[e] + bv;
          float g = 0.5f * t * (1.0f + erff(t * 0.70710678118f));  // exact gelu
          s_h[m * FFDIM + col] = (half_t)g;
        }
      }
    }
  }
  __syncthreads();

  // ---- FFN2: y = h @ w2 + b2 (WMMA, K=512, 16 A-frags register-resident) ----
  gemm_store_f32<16, 8>(s_h, FFDIM, w2p, s_tmp, DMODEL, b2, wave, lane);
  __syncthreads();

  // ---- out = LN(xr + ffn) -> global ----
  layernorm_rows(s_tmp, DMODEL, s_x32, DMODEL, /*inside=*/true, n2_w, n2_b,
                 nullptr, nullptr, outg, wave, lane);
}

// ---------------------------------------------------------------------------
// Host launch
// ---------------------------------------------------------------------------
extern "C" void kernel_launch(void* const* d_in, const int* in_sizes, int n_in,
                              void* d_out, int out_size, void* d_ws, size_t ws_size,
                              hipStream_t stream) {
  (void)in_sizes; (void)n_in; (void)out_size; (void)ws_size;
  const float* x     = (const float*)d_in[0];
  const float* wq    = (const float*)d_in[1];
  const float* wk    = (const float*)d_in[2];
  const float* wv    = (const float*)d_in[3];
  const float* lnq_w = (const float*)d_in[4];
  const float* lnq_b = (const float*)d_in[5];
  const float* lnk_w = (const float*)d_in[6];
  const float* lnk_b = (const float*)d_in[7];
  const float* lnv_w = (const float*)d_in[8];
  const float* lnv_b = (const float*)d_in[9];
  const float* rel_t = (const float*)d_in[10];
  const float* gbias = (const float*)d_in[11];
  const float* alpha = (const float*)d_in[12];
  const float* wo    = (const float*)d_in[13];
  const float* bo    = (const float*)d_in[14];
  const float* w1    = (const float*)d_in[15];
  const float* b1    = (const float*)d_in[16];
  const float* w2    = (const float*)d_in[17];
  const float* b2    = (const float*)d_in[18];
  const float* n1_w  = (const float*)d_in[19];
  const float* n1_b  = (const float*)d_in[20];
  const float* n2_w  = (const float*)d_in[21];
  const float* n2_b  = (const float*)d_in[22];
  float* outp = (float*)d_out;

  // f16 weight packs in workspace (589,824 bytes total)
  half_t* wsp   = (half_t*)d_ws;
  half_t* wpack = wsp;                 // 3*3*128*128 = 147456 halves
  half_t* wop   = wsp + 147456;        // 16384
  half_t* w1p   = wsp + 163840;        // 65536
  half_t* w2p   = wsp + 229376;        // 65536

  pack_conv_k<<<(147456 + 255) / 256, 256, 0, stream>>>(wq, wk, wv, wpack);
  pack_T_k<<<(16384 + 255) / 256, 256, 0, stream>>>(wo, wop, 128, 128);
  pack_T_k<<<(65536 + 255) / 256, 256, 0, stream>>>(w1, w1p, 128, 512);
  pack_T_k<<<(65536 + 255) / 256, 256, 0, stream>>>(w2, w2p, 512, 128);

  fused_encoder_k<<<BATCH, 256, 0, stream>>>(
      x, wpack, wop, w1p, w2p,
      lnq_w, lnq_b, lnk_w, lnk_b, lnv_w, lnv_b,
      rel_t, gbias, alpha, bo, b1, b2,
      n1_w, n1_b, n2_w, n2_b, outp);
}
